// OptimizedSparseMatchingNetwork_4277787426933
// MI455X (gfx1250) — matrix operand order repaired and checked by reading, not compiled, not run on hardware
//
#include <hip/hip_runtime.h>
#include <cmath>
#include <cstdint>

// ---------------------------------------------------------------------------
// Types / constants
// ---------------------------------------------------------------------------
typedef __attribute__((ext_vector_type(16))) __bf16 v16bf;
typedef __attribute__((ext_vector_type(8)))  float  v8f;
typedef __attribute__((ext_vector_type(4)))  unsigned int v4u;
typedef __attribute__((ext_vector_type(8)))  int    v8i;
typedef __attribute__((ext_vector_type(4)))  int    v4i;

union Frag { v16bf v; uint4 q[2]; };

constexpr int CB  = 2;          // batch
constexpr int CN  = 4096;       // tokens
constexpr int CD  = 768;        // dim
constexpr int CH  = 8;          // heads
constexpr int CHD = 96;         // head dim
constexpr int CL  = 4;          // layers
constexpr int CW  = 16;         // window
constexpr int CNW = CN / CW;    // windows per batch
constexpr int CKT = 409;        // top-k (int(4096*0.1))
constexpr int CKP = 416;        // top-k padded to 16
constexpr int TPANEL = 128;     // K-panel staged into LDS by the TDM
constexpr float RSH  = 0.10206207261596575f;   // 96^-0.5
constexpr float NEGF = -3.4028234663852886e38f;

__device__ __forceinline__ v8f wmma_bf(const Frag& a, const Frag& b, v8f c) {
  return __builtin_amdgcn_wmma_f32_16x16x32_bf16(false, a.v, false, b.v,
                                                 (short)0, c, false, false);
}

// ---------------------------------------------------------------------------
// Tensor Data Mover: DMA a [tile_r=16 x tile_c=128] bf16 tile from a row-major
// global tensor (row stride = stride0 elements) into LDS at lds_addr.
// D# built per CDNA5 ISA ch.8: group0 = {count=1, lds_addr, global_addr,
// type=2}; group1 = {data_size=2B, tensor_dim0/1 (OOB rows/cols -> zero-fill),
// tile_dim0=128, tile_dim1=16, tensor_dim0_stride}.  Tracked by TENSORcnt.
// ---------------------------------------------------------------------------
__device__ __forceinline__ void tdm_load_2d_bf16(unsigned lds_addr, const void* gptr,
                                                 unsigned tdim0, unsigned tdim1,
                                                 unsigned stride0)
{
  unsigned long long ga = (unsigned long long)gptr;
  v4u g0;
  g0[0] = 1u;                                            // count=1 (valid D#)
  g0[1] = lds_addr;                                      // LDS byte address
  g0[2] = (unsigned)(ga & 0xffffffffull);                // global_addr[31:0]
  g0[3] = (unsigned)((ga >> 32) & 0x01ffffffull) | (2u << 30);  // [56:32]|type=2
  v8i g1;
  g1[0] = (int)(1u << 16);                               // data_size=1 (2B)
  g1[1] = (int)((tdim0 & 0xffffu) << 16);                // tensor_dim0 lo
  g1[2] = (int)((tdim0 >> 16) | ((tdim1 & 0xffffu) << 16)); // dim0 hi | dim1 lo
  g1[3] = (int)((tdim1 >> 16) | ((unsigned)TPANEL << 16));  // dim1 hi | tile_dim0
  g1[4] = 16;                                            // tile_dim1=16
  g1[5] = (int)stride0;                                  // dim0_stride[31:0]
  g1[6] = 0;
  g1[7] = 0;
  v4i zz; zz[0] = 0; zz[1] = 0; zz[2] = 0; zz[3] = 0;
#if defined(__clang_major__) && (__clang_major__ >= 23)
  v8i z8;
  z8[0]=0; z8[1]=0; z8[2]=0; z8[3]=0; z8[4]=0; z8[5]=0; z8[6]=0; z8[7]=0;
  __builtin_amdgcn_tensor_load_to_lds(g0, g1, zz, zz, z8, 0);
#else
  __builtin_amdgcn_tensor_load_to_lds(g0, g1, zz, zz, 0);
#endif
}

// ---------------------------------------------------------------------------
// Batched WMMA GEMM with TDM-staged, double-buffered A panels in LDS.
//   BT=true : C[z](MxN) = A[z](MxK) * B[z](NxK)^T
//   BT=false: C[z](MxN) = A[z](MxK) * B[z](KxN)
// One block = 8 waves = 16(M) x 128(N) tile; the 16xTPANEL A panel is DMA'd to
// LDS once per block (TDM, async, double buffered) and shared by all 8 waves;
// B streams from global as b128 per-lane loads (BT) or strided scalars (NN).
// K must be a multiple of 32 (96/416/768/3072/4096 all qualify).
// ---------------------------------------------------------------------------
template<bool BT>
__global__ __launch_bounds__(256)
void k_gemm(const __bf16* __restrict__ A, long long lda, long long sa1, long long sa2,
            const __bf16* __restrict__ Bm, long long ldb, long long sb1, long long sb2,
            float* __restrict__ C, long long ldc, long long sc1, long long sc2,
            __bf16* __restrict__ Cb, long long ldcb, long long scb1, long long scb2,
            const float* __restrict__ bias, float alpha, int relu,
            int M, int Nn, int K, int Z2)
{
  __shared__ __bf16 tA[2][16][TPANEL];

  int z  = blockIdx.z;
  int z1 = z / Z2, z2 = z % Z2;
  int wid  = threadIdx.x >> 5;
  int lane = threadIdx.x & 31;
  int row  = lane & 15;
  int half = lane >> 4;
  int n0 = (blockIdx.x * 8 + wid) * 16;
  int m0 = blockIdx.y * 16;
  bool wactive = (n0 < Nn);        // idle waves still participate in barriers

  const __bf16* Az = A  + z1 * sa1 + z2 * sa2 + (long long)m0 * lda;
  const __bf16* Bz = Bm + z1 * sb1 + z2 * sb2;
  int  col = row;
  bool bn  = wactive && ((n0 + col) < Nn);
  const __bf16* brow = BT ? (Bz + (long long)(n0 + col) * ldb)
                          : (Bz + (n0 + col));

  unsigned lds0 = (unsigned)(unsigned long long)(void*)&tA[0][0][0];
  unsigned lds1 = (unsigned)(unsigned long long)(void*)&tA[1][0][0];
  unsigned mrem = (unsigned)(M - m0);

  if (wid == 0)
    tdm_load_2d_bf16(lds0, Az, (unsigned)K, mrem, (unsigned)lda);
  __builtin_amdgcn_s_wait_tensorcnt(0);
  __syncthreads();

  v8f acc;
#pragma unroll
  for (int i = 0; i < 8; ++i) acc[i] = 0.f;

  int cur = 0;
  for (int kb = 0; kb < K; kb += TPANEL) {
    // prefetch next A panel into the other buffer while computing this one
    if (wid == 0 && (kb + TPANEL) < K)
      tdm_load_2d_bf16(cur ? lds0 : lds1, Az + kb + TPANEL,
                       (unsigned)(K - kb - TPANEL), mrem, (unsigned)lda);
    if (wactive) {
      int kend = K - kb; if (kend > TPANEL) kend = TPANEL;
      const __bf16* ar = &tA[cur][row][0];
      for (int kc = 0; kc < kend; kc += 32) {
        Frag a, b;
        const __bf16* pa = ar + kc + half * 16;      // ds_load_b128 x2
        a.q[0] = *(const uint4*)(pa);
        a.q[1] = *(const uint4*)(pa + 8);
        if (BT) {
          if (bn) {
            const __bf16* pb = brow + kb + kc + half * 16;
            b.q[0] = *(const uint4*)(pb);
            b.q[1] = *(const uint4*)(pb + 8);
            __builtin_prefetch(brow + kb + kc + 256, 0, 1);
          } else {
            b.q[0] = make_uint4(0u, 0u, 0u, 0u);
            b.q[1] = make_uint4(0u, 0u, 0u, 0u);
          }
        } else {
#pragma unroll
          for (int i = 0; i < 16; ++i) {
            int kk = kb + kc + half * 16 + i;
            b.v[i] = bn ? brow[(long long)kk * ldb] : (__bf16)0.0f;
          }
        }
        acc = wmma_bf(a, b, acc);
      }
    }
    __builtin_amdgcn_s_wait_tensorcnt(0);   // prefetched panel has landed
    __syncthreads();                        // everyone done reading cur panel
    cur ^= 1;
  }

  if (wactive) {
    int n = n0 + col;
    float bv = (bias && n < Nn) ? bias[n] : 0.f;
#pragma unroll
    for (int j = 0; j < 8; ++j) {
      int m = m0 + j + half * 8;
      if (m < M && n < Nn) {
        float v = acc[j] * alpha + bv;
        if (relu) v = fmaxf(v, 0.f);
        if (C)  C [z1 * sc1  + z2 * sc2  + (long long)m * ldc  + n] = v;
        if (Cb) Cb[z1 * scb1 + z2 * scb2 + (long long)m * ldcb + n] = (__bf16)v;
      }
    }
  }
}

// ---------------------------------------------------------------------------
// Fused local windowed attention: one wave per (b, window, head).
// S = (Q K^T) * 96^-0.5 ; softmax over keys (in-register, shfl_xor over the
// 16-lane column groups of the C-fragment layout) ; O = P V via LDS re-stripe.
// ---------------------------------------------------------------------------
__global__ void k_local_attn(const __bf16* __restrict__ Q, const __bf16* __restrict__ Kx,
                             const __bf16* __restrict__ V, __bf16* __restrict__ O)
{
  __shared__ __bf16 lp[8][16 * 16 + 8];
  int wid  = threadIdx.x >> 5;
  int lane = threadIdx.x & 31;
  int gw   = blockIdx.x * 8 + wid;
  int b    = gw / (CNW * CH);
  int r    = gw % (CNW * CH);
  int win  = r / CH;
  int h    = r % CH;
  int t0   = win * CW;
  int row  = lane & 15;
  int half = lane >> 4;

  long long abase = ((long long)(b * CN + t0 + row)) * CD + h * CHD;

  v8f s;
#pragma unroll
  for (int i = 0; i < 8; ++i) s[i] = 0.f;

  for (int kc = 0; kc < CHD; kc += 32) {
    Frag a, bk;
    const __bf16* pa = Q + abase + kc + half * 16;
    a.q[0]  = *(const uint4*)(pa);
    a.q[1]  = *(const uint4*)(pa + 8);
    const __bf16* pb = Kx + abase + kc + half * 16;   // col token == row
    bk.q[0] = *(const uint4*)(pb);
    bk.q[1] = *(const uint4*)(pb + 8);
    s = wmma_bf(a, bk, s);
  }

  // softmax over key axis (columns => across the 16-lane group)
#pragma unroll
  for (int j = 0; j < 8; ++j) {
    float v = s[j] * RSH;
    float m = v;
    for (int o = 1; o < 16; o <<= 1) m = fmaxf(m, __shfl_xor(m, o, 32));
    float e = __expf(v - m);
    float su = e;
    for (int o = 1; o < 16; o <<= 1) su += __shfl_xor(su, o, 32);
    float p = e / su;
    int mrow = j + half * 8;               // query row in window
    lp[wid][mrow * 16 + row] = (__bf16)p;  // row == key col here
  }
  __syncthreads();

  // re-stripe P into an A-fragment (K padded 16 -> 32 with zeros)
  Frag ap;
#pragma unroll
  for (int i = 0; i < 16; ++i) {
    int kk = half * 16 + i;
    ap.v[i] = (kk < 16) ? lp[wid][row * 16 + kk] : (__bf16)0.0f;
  }
  __syncthreads();

  // O = P @ V over six 16-wide d tiles
#pragma unroll
  for (int dt = 0; dt < 6; ++dt) {
    Frag bv;
#pragma unroll
    for (int i = 0; i < 16; ++i) {
      int kk = half * 16 + i;              // key token index (pad >=16 -> 0)
      bv.v[i] = (kk < CW)
        ? V[((long long)(b * CN + t0 + kk)) * CD + h * CHD + dt * 16 + row]
        : (__bf16)0.0f;
    }
    v8f o;
#pragma unroll
    for (int i = 0; i < 8; ++i) o[i] = 0.f;
    o = wmma_bf(ap, bv, o);
#pragma unroll
    for (int j = 0; j < 8; ++j) {
      int m = j + half * 8;
      O[((long long)(b * CN + t0 + m)) * CD + h * CHD + dt * 16 + row] = (__bf16)o[j];
    }
  }
}

// ---------------------------------------------------------------------------
// Elementwise / reduction kernels
// ---------------------------------------------------------------------------
__global__ void k_add_ln(const float* __restrict__ X, const float* __restrict__ P,
                         const float* __restrict__ G, const float* __restrict__ Bv,
                         float* __restrict__ Y, __bf16* __restrict__ Yb)
{
  __shared__ float red[256];
  int t = threadIdx.x;
  long long rbase = (long long)blockIdx.x * CD;
  float tv[3];
  float s = 0.f;
#pragma unroll
  for (int i = 0; i < 3; ++i) {
    int c = t + i * 256;
    float v = X[rbase + c] + (P ? P[rbase + c] : 0.f);
    tv[i] = v; s += v;
  }
  red[t] = s; __syncthreads();
  for (int o = 128; o > 0; o >>= 1) { if (t < o) red[t] += red[t + o]; __syncthreads(); }
  float mean = red[0] / (float)CD; __syncthreads();
  float s2 = 0.f;
#pragma unroll
  for (int i = 0; i < 3; ++i) { float d = tv[i] - mean; s2 += d * d; }
  red[t] = s2; __syncthreads();
  for (int o = 128; o > 0; o >>= 1) { if (t < o) red[t] += red[t + o]; __syncthreads(); }
  float rs = rsqrtf(red[0] / (float)CD + 1e-5f);
#pragma unroll
  for (int i = 0; i < 3; ++i) {
    int c = t + i * 256;
    float y = (tv[i] - mean) * rs * G[c] + Bv[c];
    if (Y)  Y[rbase + c]  = y;
    if (Yb) Yb[rbase + c] = (__bf16)y;
  }
}

__global__ void k_rowsumsq(const float* __restrict__ X, float* __restrict__ IMP)
{
  __shared__ float red[256];
  int t = threadIdx.x;
  long long rbase = (long long)blockIdx.x * CD;
  float s = 0.f;
#pragma unroll
  for (int i = 0; i < 3; ++i) { float v = X[rbase + t + i * 256]; s += v * v; }
  red[t] = s; __syncthreads();
  for (int o = 128; o > 0; o >>= 1) { if (t < o) red[t] += red[t + o]; __syncthreads(); }
  if (t == 0) IMP[blockIdx.x] = red[0];   // sqrt is monotonic; skip for top-k
}

__global__ __launch_bounds__(1024)
void k_topk(const float* __restrict__ IMP, int* __restrict__ IDX)
{
  __shared__ float vals[CN];
  __shared__ float rv[1024];
  __shared__ int   ri[1024];
  int t = threadIdx.x;
  for (int i = t; i < CN; i += 1024) vals[i] = IMP[(long long)blockIdx.x * CN + i];
  __syncthreads();
  for (int pass = 0; pass < CKT; ++pass) {
    float bv = NEGF; int bi = CN;
    for (int i = t; i < CN; i += 1024) {
      float v = vals[i];
      if (v > bv || (v == bv && i < bi)) { bv = v; bi = i; }
    }
    rv[t] = bv; ri[t] = bi; __syncthreads();
    for (int o = 512; o > 0; o >>= 1) {
      if (t < o) {
        float ov = rv[t + o]; int oi = ri[t + o];
        if (ov > rv[t] || (ov == rv[t] && oi < ri[t])) { rv[t] = ov; ri[t] = oi; }
      }
      __syncthreads();
    }
    if (t == 0) {
      IDX[(long long)blockIdx.x * CKT + pass] = ri[0];
      vals[ri[0]] = NEGF;
    }
    __syncthreads();
  }
}

__global__ void k_gather(const float* __restrict__ X, const int* __restrict__ IDX,
                         __bf16* __restrict__ OUT)
{
  int b = blockIdx.x / CKP, j = blockIdx.x % CKP;
  long long dbase = (long long)blockIdx.x * CD;
  if (j < CKT) {
    int r = IDX[(long long)b * CKT + j];
    long long sbase = ((long long)b * CN + r) * CD;
#pragma unroll
    for (int i = 0; i < 3; ++i) {
      int c = threadIdx.x + i * 256;
      OUT[dbase + c] = (__bf16)X[sbase + c];
    }
  } else {
#pragma unroll
    for (int i = 0; i < 3; ++i) OUT[dbase + threadIdx.x + i * 256] = (__bf16)0.0f;
  }
}

__global__ void k_scatter(float* __restrict__ DST, const float* __restrict__ SRC,
                          const int* __restrict__ IDX)
{
  int b = blockIdx.x / CKT, j = blockIdx.x % CKT;
  int r = IDX[(long long)b * CKT + j];
  long long dbase = ((long long)b * CN + r) * CD;
  long long sbase = ((long long)b * CKP + j) * CD;
#pragma unroll
  for (int i = 0; i < 3; ++i) {
    int c = threadIdx.x + i * 256;
    DST[dbase + c] = SRC[sbase + c];
  }
}

__global__ void k_add(const float* __restrict__ A, const float* __restrict__ Bv,
                      float* __restrict__ C, long long n)
{
  long long stride = (long long)gridDim.x * blockDim.x;
  for (long long i = (long long)blockIdx.x * blockDim.x + threadIdx.x; i < n; i += stride)
    C[i] = A[i] + Bv[i];
}

__global__ void k_pack(const float* __restrict__ X, __bf16* __restrict__ Y, long long n)
{
  long long stride = (long long)gridDim.x * blockDim.x;
  for (long long i = (long long)blockIdx.x * blockDim.x + threadIdx.x; i < n; i += stride)
    Y[i] = (__bf16)X[i];
}

__global__ void k_pack_wT(const float* __restrict__ W, __bf16* __restrict__ Wt,
                          int Kd, int Nd)
{
  long long tot = (long long)Kd * Nd;
  long long stride = (long long)gridDim.x * blockDim.x;
  for (long long i = (long long)blockIdx.x * blockDim.x + threadIdx.x; i < tot; i += stride) {
    long long k = i / Nd;
    int n = (int)(i % Nd);
    Wt[(long long)n * Kd + k] = (__bf16)W[i];
  }
}

__global__ void k_softmax_bf(const float* __restrict__ S, __bf16* __restrict__ P,
                             int cols, int valid, float scale)
{
  __shared__ float red[256];
  int t = threadIdx.x;
  long long base = (long long)blockIdx.x * cols;
  float mx = NEGF;
  for (int c = t; c < valid; c += 256) mx = fmaxf(mx, S[base + c] * scale);
  red[t] = mx; __syncthreads();
  for (int o = 128; o > 0; o >>= 1) { if (t < o) red[t] = fmaxf(red[t], red[t + o]); __syncthreads(); }
  float m = red[0]; __syncthreads();
  float su = 0.f;
  for (int c = t; c < valid; c += 256) su += __expf(S[base + c] * scale - m);
  red[t] = su; __syncthreads();
  for (int o = 128; o > 0; o >>= 1) { if (t < o) red[t] += red[t + o]; __syncthreads(); }
  float inv = 1.f / red[0];
  for (int c = t; c < cols; c += 256) {
    float p = (c < valid) ? __expf(S[base + c] * scale - m) * inv : 0.f;
    P[base + c] = (__bf16)p;
  }
}

__global__ void k_pos_feat(const float* __restrict__ KP, const int* __restrict__ Hp,
                           const int* __restrict__ Wp, __bf16* __restrict__ OUT)
{
  float W1 = (float)(Wp[0] - 1), H1 = (float)(Hp[0] - 1);
  long long row = blockIdx.x;
  float px = KP[row * 2 + 0] / W1;
  float py = KP[row * 2 + 1] / H1;
#pragma unroll
  for (int i = 0; i < 3; ++i) {
    int e = threadIdx.x + i * 256;
    int quad = e / 192, fi = e % 192;
    float f = exp2f(fminf((float)fi, 60.f));
    float v = (quad == 0) ? __sinf(px * f)
            : (quad == 1) ? __cosf(px * f)
            : (quad == 2) ? __sinf(py * f)
                          : __cosf(py * f);
    OUT[row * CD + e] = (__bf16)v;
  }
}

__global__ void k_final(float* __restrict__ MS, float* __restrict__ DISP,
                        float* __restrict__ MK, const float* __restrict__ KPL,
                        const float* __restrict__ KPR)
{
  __shared__ float red[256];
  int t = threadIdx.x;
  int b = blockIdx.x / CN, q = blockIdx.x % CN;
  long long base = ((long long)b * CN + q) * CN;
  float xl = KPL[((long long)b * CN + q) * 2 + 0];
  float yl = KPL[((long long)b * CN + q) * 2 + 1];
  float mloc = NEGF;
  for (int m = t; m < CN; m += 256) {
    float xr = KPR[((long long)b * CN + m) * 2 + 0];
    float yr = KPR[((long long)b * CN + m) * 2 + 1];
    bool ok = (xl >= xr) && (fabsf(yl - yr) < 2.0f);
    float sv = ok ? MS[base + m] : NEGF;
    MS[base + m] = sv;
    MK[base + m] = ok ? 1.0f : 0.0f;
    mloc = fmaxf(mloc, sv * 0.5f);   // TEMP = 0.5
  }
  red[t] = mloc; __syncthreads();
  for (int o = 128; o > 0; o >>= 1) { if (t < o) red[t] = fmaxf(red[t], red[t + o]); __syncthreads(); }
  float mx = red[0]; __syncthreads();
  float se = 0.f, sd = 0.f;
  for (int m = t; m < CN; m += 256) {
    float e  = __expf(MS[base + m] * 0.5f - mx);
    float xr = KPR[((long long)b * CN + m) * 2 + 0];
    se += e; sd += (xl - xr) * e;
  }
  red[t] = se; __syncthreads();
  for (int o = 128; o > 0; o >>= 1) { if (t < o) red[t] += red[t + o]; __syncthreads(); }
  float tot = red[0]; __syncthreads();
  red[t] = sd; __syncthreads();
  for (int o = 128; o > 0; o >>= 1) { if (t < o) red[t] += red[t + o]; __syncthreads(); }
  if (t == 0) {
    float d = red[0] / tot;
    if (!(d == d)) d = 0.f;                 // nan_to_num
    DISP[(long long)b * CN + q] = fmaxf(d, 0.f);
  }
}

// ---------------------------------------------------------------------------
// Host orchestration
// ---------------------------------------------------------------------------
static inline int cdiv(int a, int b) { return (a + b - 1) / b; }

static void launch_gemm(hipStream_t st, bool bt,
                        const __bf16* A, long long lda, long long sa1, long long sa2,
                        const __bf16* Bm, long long ldb, long long sb1, long long sb2,
                        float* C, long long ldc, long long sc1, long long sc2,
                        __bf16* Cb, long long ldcb, long long scb1, long long scb2,
                        const float* bias, float alpha, int relu,
                        int M, int Nn, int K, int Z, int Z2)
{
  dim3 g((unsigned)cdiv(Nn, 128), (unsigned)cdiv(M, 16), (unsigned)Z);
  dim3 blk(256);
  if (bt)
    k_gemm<true><<<g, blk, 0, st>>>(A, lda, sa1, sa2, Bm, ldb, sb1, sb2,
                                    C, ldc, sc1, sc2, Cb, ldcb, scb1, scb2,
                                    bias, alpha, relu, M, Nn, K, Z2);
  else
    k_gemm<false><<<g, blk, 0, st>>>(A, lda, sa1, sa2, Bm, ldb, sb1, sb2,
                                     C, ldc, sc1, sc2, Cb, ldcb, scb1, scb2,
                                     bias, alpha, relu, M, Nn, K, Z2);
}

struct AP { const float *wq, *bq, *wk, *bk, *wv, *bv, *wo, *bo; };

extern "C" void kernel_launch(void* const* d_in, const int* in_sizes, int n_in,
                              void* d_out, int out_size, void* d_ws, size_t ws_size,
                              hipStream_t stream)
{
  (void)in_sizes; (void)n_in; (void)out_size; (void)ws_size;

  // --- input flattening (jax tree order: dict keys sorted) -----------------
  auto F = [&](int i) -> const float* { return (const float*)d_in[i]; };
  const float* desc_l = F(0);
  const float* desc_r = F(1);
  const float* kp_l   = F(2);
  const float* kp_r   = F(3);
  auto getAP = [&](int base) -> AP {
    AP p;
    p.bk = F(base + 0); p.bo = F(base + 1); p.bq = F(base + 2); p.bv = F(base + 3);
    p.wk = F(base + 4); p.wo = F(base + 5); p.wq = F(base + 6); p.wv = F(base + 7);
    return p;
  };
  AP cross_p = getAP(4);
  const float *ffn_b1 = F(12), *ffn_b2 = F(13), *ffn_w1 = F(14), *ffn_w2 = F(15);
  const float *lnc_b = F(16), *lnc_g = F(17);
  const float *lnf_b = F(18), *lnf_g = F(19);
  const float *lnl_b = F(20), *lnl_g = F(21);
  const float *lnr_b = F(22), *lnr_g = F(23);
  const float *pos_b = F(24), *pos_w = F(25);
  const int* Hp = (const int*)d_in[154];
  const int* Wp = (const int*)d_in[155];
  // sa_left layer i: global @ 26+i*16, local @ 26+i*16+8 ; sa_right base 90

  // --- workspace bump allocator -------------------------------------------
  size_t off = 0;
  auto alloc = [&](size_t bytes) -> void* {
    void* p = (char*)d_ws + off;
    off += (bytes + 255) & ~(size_t)255;
    return p;
  };
  const long long SZ  = (long long)CB * CN * CD;
  const long long GSZ = (long long)CB * CKP * CD;

  float*  fl    = (float*)alloc(SZ * 4);
  float*  fr    = (float*)alloc(SZ * 4);
  float*  posL  = (float*)alloc(SZ * 4);
  float*  posR  = (float*)alloc(SZ * 4);
  float*  xf    = (float*)alloc(SZ * 4);
  float*  projf = (float*)alloc(SZ * 4);
  float*  gout  = (float*)alloc(GSZ * 4);
  float*  imp   = (float*)alloc((long long)CB * CN * 4);
  int*    idxb  = (int*)  alloc((long long)CB * CKT * 4);
  __bf16* xbf   = (__bf16*)alloc(SZ * 2);
  __bf16* frpk  = (__bf16*)alloc(SZ * 2);
  __bf16* qbf   = (__bf16*)alloc(SZ * 2);
  __bf16* kbf   = (__bf16*)alloc(SZ * 2);
  __bf16* vbf   = (__bf16*)alloc(SZ * 2);
  __bf16* hObf  = (__bf16*)alloc(SZ * 2);
  __bf16* xgbf  = (__bf16*)alloc(GSZ * 2);
  __bf16* qg    = (__bf16*)alloc(GSZ * 2);
  __bf16* kg    = (__bf16*)alloc(GSZ * 2);
  __bf16* vg    = (__bf16*)alloc(GSZ * 2);
  __bf16* gObf  = (__bf16*)alloc(GSZ * 2);
  float*  sg    = (float*) alloc((long long)CB * CH * CKP * CKP * 4);
  __bf16* pg    = (__bf16*)alloc((long long)CB * CH * CKP * CKP * 2);
  float*  sbuf  = (float*) alloc((long long)CN * CN * 4);
  __bf16* pbuf  = (__bf16*)alloc((long long)CN * CN * 2);
  __bf16* h1bf  = (__bf16*)alloc((long long)CB * CN * 4 * CD * 2);
  __bf16* wtbuf = (__bf16*)alloc((long long)CD * 4 * CD * 2);

  float* ms_out   = (float*)d_out;
  float* disp_out = ms_out + (long long)CB * CN * CN;
  float* mask_out = disp_out + (long long)CB * CN;

  // pack a weight (K x N f32) transposed into wtbuf, then GEMM from it
  auto proj = [&](const __bf16* Abf, int M, const float* W, const float* bias,
                  __bf16* outBf, float* outF, int Kd, int Nd, int relu) {
    k_pack_wT<<<1024, 256, 0, stream>>>(W, wtbuf, Kd, Nd);
    launch_gemm(stream, true, Abf, Kd, 0, 0, wtbuf, Kd, 0, 0,
                outF, Nd, 0, 0, outBf, Nd, 0, 0, bias, 1.f, relu,
                M, Nd, Kd, 1, 1);
  };

  // --- init: fl/fr = descriptors, positional encodings --------------------
  hipMemcpyAsync(fl, desc_l, SZ * 4, hipMemcpyDeviceToDevice, stream);
  hipMemcpyAsync(fr, desc_r, SZ * 4, hipMemcpyDeviceToDevice, stream);

  k_pos_feat<<<CB * CN, 256, 0, stream>>>(kp_l, Hp, Wp, xbf);
  proj(xbf, CB * CN, pos_w, pos_b, nullptr, posL, CD, CD, 0);
  k_pos_feat<<<CB * CN, 256, 0, stream>>>(kp_r, Hp, Wp, xbf);
  proj(xbf, CB * CN, pos_w, pos_b, nullptr, posR, CD, CD, 0);

  // --- one sparse-attention layer for one side -----------------------------
  auto run_layer_side = [&](float* f, const float* pos, const float* lng,
                            const float* lnb, AP gp, AP lo) {
    k_add_ln<<<CB * CN, 256, 0, stream>>>(f, pos, lng, lnb, xf, xbf);
    k_rowsumsq<<<CB * CN, 256, 0, stream>>>(xf, imp);
    k_topk<<<CB, 1024, 0, stream>>>(imp, idxb);
    k_gather<<<CB * CKP, 256, 0, stream>>>(xf, idxb, xgbf);

    // global attention over top-k tokens (padded to 416)
    proj(xgbf, CB * CKP, gp.wq, gp.bq, qg, nullptr, CD, CD, 0);
    proj(xgbf, CB * CKP, gp.wk, gp.bk, kg, nullptr, CD, CD, 0);
    proj(xgbf, CB * CKP, gp.wv, gp.bv, vg, nullptr, CD, CD, 0);
    launch_gemm(stream, true,
                qg, CD, (long long)CKP * CD, CHD,
                kg, CD, (long long)CKP * CD, CHD,
                sg, CKP, (long long)CH * CKP * CKP, (long long)CKP * CKP,
                nullptr, 0, 0, 0, nullptr, RSH, 0,
                CKP, CKP, CHD, CB * CH, CH);
    k_softmax_bf<<<CB * CH * CKP, 256, 0, stream>>>(sg, pg, CKP, CKT, 1.f);
    launch_gemm(stream, false,
                pg, CKP, (long long)CH * CKP * CKP, (long long)CKP * CKP,
                vg, CD, (long long)CKP * CD, CHD,
                nullptr, 0, 0, 0,
                gObf, CD, (long long)CKP * CD, CHD,
                nullptr, 1.f, 0, CKP, CHD, CKP, CB * CH, CH);
    proj(gObf, CB * CKP, gp.wo, gp.bo, nullptr, gout, CD, CD, 0);

    // local windowed attention (all tokens)
    proj(xbf, CB * CN, lo.wq, lo.bq, qbf, nullptr, CD, CD, 0);
    proj(xbf, CB * CN, lo.wk, lo.bk, kbf, nullptr, CD, CD, 0);
    proj(xbf, CB * CN, lo.wv, lo.bv, vbf, nullptr, CD, CD, 0);
    k_local_attn<<<CB * CNW * CH / 8, 256, 0, stream>>>(qbf, kbf, vbf, hObf);
    proj(hObf, CB * CN, lo.wo, lo.bo, nullptr, projf, CD, CD, 0);

    // combine: local rows at top-k indices replaced by global rows, residual
    k_scatter<<<CB * CKT, 256, 0, stream>>>(projf, gout, idxb);
    k_add<<<4096, 256, 0, stream>>>(f, projf, f, SZ);
  };

  for (int layer = 0; layer < CL; ++layer) {
    run_layer_side(fl, posL, lnl_g, lnl_b,
                   getAP(26 + layer * 16), getAP(26 + layer * 16 + 8));
    run_layer_side(fr, posR, lnr_g, lnr_b,
                   getAP(90 + layer * 16), getAP(90 + layer * 16 + 8));
  }

  // --- cross attention: Q from fl, K/V from fr -----------------------------
  k_pack<<<4096, 256, 0, stream>>>(fl, xbf, SZ);
  k_pack<<<4096, 256, 0, stream>>>(fr, frpk, SZ);
  proj(xbf,  CB * CN, cross_p.wq, cross_p.bq, qbf, nullptr, CD, CD, 0);
  proj(frpk, CB * CN, cross_p.wk, cross_p.bk, kbf, nullptr, CD, CD, 0);
  proj(frpk, CB * CN, cross_p.wv, cross_p.bv, vbf, nullptr, CD, CD, 0);
  for (int b = 0; b < CB; ++b) {
    for (int h = 0; h < CH; ++h) {
      long long hb = (long long)b * CN * CD + h * CHD;
      launch_gemm(stream, true, qbf + hb, CD, 0, 0, kbf + hb, CD, 0, 0,
                  sbuf, CN, 0, 0, nullptr, 0, 0, 0,
                  nullptr, RSH, 0, CN, CN, CHD, 1, 1);
      k_softmax_bf<<<CN, 256, 0, stream>>>(sbuf, pbuf, CN, CN, 1.f);
      launch_gemm(stream, false, pbuf, CN, 0, 0, vbf + hb, CD, 0, 0,
                  nullptr, 0, 0, 0, hObf + hb, CD, 0, 0,
                  nullptr, 1.f, 0, CN, CHD, CN, 1, 1);
    }
  }
  proj(hObf, CB * CN, cross_p.wo, cross_p.bo, nullptr, projf, CD, CD, 0);

  // fle = fl + LN(cross)
  k_add_ln<<<CB * CN, 256, 0, stream>>>(projf, nullptr, lnc_g, lnc_b, xf, nullptr);
  k_add<<<4096, 256, 0, stream>>>(fl, xf, fl, SZ);

  // FFN with residual
  k_add_ln<<<CB * CN, 256, 0, stream>>>(fl, nullptr, lnf_g, lnf_b, nullptr, xbf);
  proj(xbf, CB * CN, ffn_w1, ffn_b1, h1bf, nullptr, CD, 4 * CD, 1);   // relu
  proj(h1bf, CB * CN, ffn_w2, ffn_b2, nullptr, projf, 4 * CD, CD, 0);
  k_add<<<4096, 256, 0, stream>>>(fl, projf, fl, SZ);

  // --- match scores: ms = fle @ fr^T, then mask/softmax/disparity ----------
  k_pack<<<4096, 256, 0, stream>>>(fl, xbf, SZ);
  k_pack<<<4096, 256, 0, stream>>>(fr, frpk, SZ);
  launch_gemm(stream, true,
              xbf, CD, (long long)CN * CD, 0,
              frpk, CD, (long long)CN * CD, 0,
              ms_out, CN, (long long)CN * CN, 0,
              nullptr, 0, 0, 0,
              nullptr, 1.f, 0, CN, CN, CD, CB, 1);
  k_final<<<CB * CN, 256, 0, stream>>>(ms_out, disp_out, mask_out, kp_l, kp_r);
}
